// Block_6167573037529
// MI455X (gfx1250) — matrix operand order repaired and checked by reading, not compiled, not run on hardware
//
#include <hip/hip_runtime.h>

// ---------------------------------------------------------------------------
// Types
// ---------------------------------------------------------------------------
typedef __attribute__((ext_vector_type(16))) __bf16 v16bf;
typedef __attribute__((ext_vector_type(8)))  __bf16 v8bf;
typedef __attribute__((ext_vector_type(8)))  float  v8f;

static __device__ inline v16bf combine16(v8bf lo, v8bf hi) {
    return __builtin_shufflevector(lo, hi, 0,1,2,3,4,5,6,7,8,9,10,11,12,13,14,15);
}

static __device__ inline v8f wmma_bf16(v16bf a, v16bf b, v8f c) {
    // v_wmma_f32_16x16x32_bf16: D = A(16x32) * B(32x16) + C(16x16 f32)
    return __builtin_amdgcn_wmma_f32_16x16x32_bf16(false, a, false, b, (short)0, c, false, false);
}

static __device__ inline float gelu_tanh(float x) {
    float x3 = x * x * x;
    return 0.5f * x * (1.0f + tanhf(0.7978845608028654f * (x + 0.044715f * x3)));
}

// ---------------------------------------------------------------------------
// fp32 -> bf16 conversion (weights)
// ---------------------------------------------------------------------------
__global__ __launch_bounds__(256) void cvt_bf16_kernel(const float* __restrict__ in,
                                                       __bf16* __restrict__ out, int n) {
    int i = blockIdx.x * 256 + threadIdx.x;
    if (i < n) out[i] = (__bf16)in[i];
}

// ---------------------------------------------------------------------------
// LayerNorm: one row (C=1024) per block, bf16 output
// ---------------------------------------------------------------------------
__global__ __launch_bounds__(256) void ln_kernel(const float* __restrict__ x,
                                                 const float* __restrict__ g,
                                                 const float* __restrict__ b,
                                                 __bf16* __restrict__ out, int C) {
    __shared__ float red[256];
    const int row = blockIdx.x;
    const float* xr = x + (size_t)row * C;
    float s = 0.f, s2 = 0.f;
    for (int i = threadIdx.x; i < C; i += 256) { float v = xr[i]; s += v; s2 += v * v; }
    red[threadIdx.x] = s; __syncthreads();
    for (int o = 128; o > 0; o >>= 1) { if (threadIdx.x < o) red[threadIdx.x] += red[threadIdx.x + o]; __syncthreads(); }
    float mean = red[0] / (float)C; __syncthreads();
    red[threadIdx.x] = s2; __syncthreads();
    for (int o = 128; o > 0; o >>= 1) { if (threadIdx.x < o) red[threadIdx.x] += red[threadIdx.x + o]; __syncthreads(); }
    float var = red[0] / (float)C - mean * mean;
    float rstd = rsqrtf(var + 1e-5f);
    __bf16* orow = out + (size_t)row * C;
    for (int i = threadIdx.x; i < C; i += 256)
        orow[i] = (__bf16)((xr[i] - mean) * rstd * g[i] + b[i]);
}

// ---------------------------------------------------------------------------
// Tiled bf16 GEMM with WMMA.  out = act(A[M,K] @ B[K,N] + bias) (+ resid)
// Block tile 128x64, 8 waves (4M x 2N), each wave 32x32 (2x2 WMMA tiles), BK=32.
// ---------------------------------------------------------------------------
#define BM 128
#define BN 64
#define BK 32

__global__ __launch_bounds__(256, 2) void gemm_bf16_kernel(
    const __bf16* __restrict__ A,     // [M,K]
    const __bf16* __restrict__ Bw,    // [K,N]
    const float*  __restrict__ bias,  // [N]
    const float*  __restrict__ resid, // [M,N] or null
    float*        __restrict__ outF,  // [M,N] or null
    __bf16*       __restrict__ outB,  // [M,N] or null
    int M, int N, int K, int act_gelu)
{
    __shared__ __attribute__((aligned(16))) __bf16 sA[BM][BK];   // 8 KB
    __shared__ __attribute__((aligned(16))) __bf16 sBt[BN][BK];  // 4 KB, B transposed

    const int tid  = threadIdx.x;
    const int lane = tid & 31;
    const int wave = tid >> 5;
    const int wm   = wave >> 1;         // 0..3
    const int wn   = wave & 1;          // 0..1
    const int hi   = lane >> 4;         // half-wave select
    const int l15  = lane & 15;
    const int bm0  = blockIdx.y * BM;
    const int bn0  = blockIdx.x * BN;

    v8f acc[2][2];
    #pragma unroll
    for (int mt = 0; mt < 2; ++mt)
        #pragma unroll
        for (int nt = 0; nt < 2; ++nt)
            acc[mt][nt] = (v8f){0.f,0.f,0.f,0.f,0.f,0.f,0.f,0.f};

    for (int k0 = 0; k0 < K; k0 += BK) {
        __syncthreads();
        // --- stage A tile 128x32 (row major, contiguous 32B per thread) ---
        {
            int r = tid >> 1, c = (tid & 1) * 16;
            const __bf16* gp = A + (size_t)(bm0 + r) * K + k0 + c;
            *(v8bf*)&sA[r][c]     = *(const v8bf*)gp;
            *(v8bf*)&sA[r][c + 8] = *(const v8bf*)(gp + 8);
        }
        // --- stage B tile 32x64 transposed -> sBt[n][k] ---
        {
            int k  = tid >> 3;           // 0..31
            int n0 = (tid & 7) * 8;      // 0..56
            v8bf v = *(const v8bf*)(Bw + (size_t)(k0 + k) * N + bn0 + n0);
            #pragma unroll
            for (int j = 0; j < 8; ++j) sBt[n0 + j][k] = v[j];
        }
        __syncthreads();

        // --- fragments ---
        v16bf afrag[2], bfrag[2];
        #pragma unroll
        for (int mt = 0; mt < 2; ++mt) {
            // 16-bit A 16x32 layout: lane<16 -> K {0..7},{16..23}; lane>=16 -> K {8..15},{24..31}
            const __bf16* p = &sA[wm * 32 + mt * 16 + l15][hi * 8];
            afrag[mt] = combine16(*(const v8bf*)p, *(const v8bf*)(p + 16));
        }
        #pragma unroll
        for (int nt = 0; nt < 2; ++nt) {
            // 16-bit B 32x16 layout: lane<16 -> K {0..7},{16..23}; lane>=16 -> K {8..15},{24..31} (col = l15)
            const __bf16* p = &sBt[wn * 32 + nt * 16 + l15][hi * 8];
            bfrag[nt] = combine16(*(const v8bf*)p, *(const v8bf*)(p + 16));
        }
        #pragma unroll
        for (int mt = 0; mt < 2; ++mt)
            #pragma unroll
            for (int nt = 0; nt < 2; ++nt)
                acc[mt][nt] = wmma_bf16(afrag[mt], bfrag[nt], acc[mt][nt]);
    }

    // --- epilogue: C layout reg i -> row = i + 8*hi, col = l15 ---
    #pragma unroll
    for (int mt = 0; mt < 2; ++mt) {
        #pragma unroll
        for (int nt = 0; nt < 2; ++nt) {
            int col = bn0 + wn * 32 + nt * 16 + l15;
            float bv = bias[col];
            #pragma unroll
            for (int i = 0; i < 8; ++i) {
                int row = bm0 + wm * 32 + mt * 16 + i + hi * 8;
                float c = acc[mt][nt][i] + bv;
                if (act_gelu) c = gelu_tanh(c);
                if (resid)    c += resid[(size_t)row * N + col];
                if (outF)     outF[(size_t)row * N + col] = c;
                if (outB)     outB[(size_t)row * N + col] = (__bf16)c;
            }
        }
    }
}

// ---------------------------------------------------------------------------
// Flash attention with WMMA.  qkv bf16 [B*T, 3C], y bf16 [B*T, C].
// Block: 128 q rows of one (b,h); 8 waves x 16 q rows; key step 32.
// ---------------------------------------------------------------------------
__global__ __launch_bounds__(256, 2) void attn_kernel(
    const __bf16* __restrict__ qkv,
    __bf16* __restrict__ y,
    int T, int C, int H, int D)
{
    const int bh   = blockIdx.x;
    const int b    = bh / H;
    const int h    = bh % H;
    const int qblk = blockIdx.y;

    const int tid  = threadIdx.x;
    const int lane = tid & 31;
    const int wave = tid >> 5;
    const int hi   = lane >> 4;
    const int l15  = lane & 15;

    const size_t row0   = (size_t)b * T;
    const int    stride = 3 * C;
    const __bf16* qp = qkv + h * D;
    const __bf16* kp = qkv + C + h * D;
    const __bf16* vp = qkv + 2 * C + h * D;

    __shared__ __attribute__((aligned(16))) __bf16 sK[32][64];       // 4 KB  [key][d]
    __shared__ __attribute__((aligned(16))) __bf16 sVt[64][32];      // 4 KB  [d][key]
    __shared__ __attribute__((aligned(16))) __bf16 sP[8][16][32];    // 8 KB  per-wave P tile

    const int qbase = qblk * 128 + wave * 16;

    // Q fragments for the two d-steps (d: 0..31, 32..63), kept in regs
    v16bf qfrag[2];
    #pragma unroll
    for (int s = 0; s < 2; ++s) {
        const __bf16* p = qp + (row0 + qbase + l15) * stride + s * 32 + hi * 8;
        qfrag[s] = combine16(*(const v8bf*)p, *(const v8bf*)(p + 16));
    }

    v8f o[4];
    #pragma unroll
    for (int nt = 0; nt < 4; ++nt) o[nt] = (v8f){0.f,0.f,0.f,0.f,0.f,0.f,0.f,0.f};
    float mrow[8], lrow[8];
    #pragma unroll
    for (int i = 0; i < 8; ++i) { mrow[i] = -1e30f; lrow[i] = 0.f; }

    const int kmax = qblk * 128 + 128;   // causal bound for whole block
    for (int kb = 0; kb < kmax; kb += 32) {
        __syncthreads();
        // cooperative stage: K tile [32 keys][64 d], V tile transposed
        {
            int key = tid >> 3, d0 = (tid & 7) * 8;
            const __bf16* gk = kp + (row0 + kb + key) * stride + d0;
            *(v8bf*)&sK[key][d0] = *(const v8bf*)gk;
            v8bf vv = *(const v8bf*)(vp + (row0 + kb + key) * stride + d0);
            #pragma unroll
            for (int j = 0; j < 8; ++j) sVt[d0 + j][key] = vv[j];
        }
        __syncthreads();

        if (kb <= qbase + 15) {   // wave-uniform causal guard -> EXEC all ones inside
            // S = Q K^T : two 16x16 tiles (keys kb..kb+15, kb+16..kb+31)
            v8f s0 = (v8f){0.f,0.f,0.f,0.f,0.f,0.f,0.f,0.f};
            v8f s1 = s0;
            #pragma unroll
            for (int s = 0; s < 2; ++s) {
                const __bf16* p0 = &sK[l15][s * 32 + hi * 8];
                const __bf16* p1 = &sK[16 + l15][s * 32 + hi * 8];
                v16bf b0 = combine16(*(const v8bf*)p0, *(const v8bf*)(p0 + 16));
                v16bf b1 = combine16(*(const v8bf*)p1, *(const v8bf*)(p1 + 16));
                s0 = wmma_bf16(qfrag[s], b0, s0);
                s1 = wmma_bf16(qfrag[s], b1, s1);
            }
            // online softmax over 32 keys (row = i + 8*hi within this q tile)
            const float sc = 0.125f;  // 1/sqrt(64)
            #pragma unroll
            for (int i = 0; i < 8; ++i) {
                float e0 = s0[i] * sc, e1 = s1[i] * sc;
                int qrow = qbase + i + hi * 8;
                if (kb + l15      > qrow) e0 = -1e30f;
                if (kb + 16 + l15 > qrow) e1 = -1e30f;
                float m = fmaxf(e0, e1);
                m = fmaxf(m, __shfl_xor(m, 1, 32));
                m = fmaxf(m, __shfl_xor(m, 2, 32));
                m = fmaxf(m, __shfl_xor(m, 4, 32));
                m = fmaxf(m, __shfl_xor(m, 8, 32));
                float nm   = fmaxf(mrow[i], m);
                float corr = __expf(mrow[i] - nm);
                mrow[i] = nm;
                float p0 = __expf(e0 - nm), p1 = __expf(e1 - nm);
                float rs = p0 + p1;
                rs += __shfl_xor(rs, 1, 32);
                rs += __shfl_xor(rs, 2, 32);
                rs += __shfl_xor(rs, 4, 32);
                rs += __shfl_xor(rs, 8, 32);
                lrow[i] = lrow[i] * corr + rs;
                #pragma unroll
                for (int nt = 0; nt < 4; ++nt) o[nt][i] *= corr;
                sP[wave][i + hi * 8][l15]      = (__bf16)p0;
                sP[wave][i + hi * 8][16 + l15] = (__bf16)p1;
            }
            // O += P(16x32) @ V(32x64) : restage P via LDS into A-fragment layout
            const __bf16* pp = &sP[wave][l15][hi * 8];
            v16bf pfrag = combine16(*(const v8bf*)pp, *(const v8bf*)(pp + 16));
            #pragma unroll
            for (int nt = 0; nt < 4; ++nt) {
                const __bf16* bp = &sVt[nt * 16 + l15][hi * 8];
                v16bf bfrag = combine16(*(const v8bf*)bp, *(const v8bf*)(bp + 16));
                o[nt] = wmma_bf16(pfrag, bfrag, o[nt]);
            }
        }
    }

    // epilogue: normalize and store y[B*T, C] (bf16, head slice)
    #pragma unroll
    for (int nt = 0; nt < 4; ++nt) {
        #pragma unroll
        for (int i = 0; i < 8; ++i) {
            int row = qbase + i + hi * 8;
            int d   = nt * 16 + l15;
            y[(row0 + row) * (size_t)C + h * D + d] = (__bf16)(o[nt][i] / lrow[i]);
        }
    }
}

// ---------------------------------------------------------------------------
// Host launcher
// ---------------------------------------------------------------------------
extern "C" void kernel_launch(void* const* d_in, const int* in_sizes, int n_in,
                              void* d_out, int out_size, void* d_ws, size_t ws_size,
                              hipStream_t stream) {
    (void)in_sizes; (void)n_in; (void)out_size; (void)ws_size;
    constexpr int Bsz = 8, T = 1024, Cn = 1024, Hn = 16, Dn = 64;
    constexpr int M = Bsz * T;

    const float* x      = (const float*)d_in[0];
    const float* ln1_g  = (const float*)d_in[1];
    const float* ln1_b  = (const float*)d_in[2];
    const float* w_qkv  = (const float*)d_in[3];
    const float* b_qkv  = (const float*)d_in[4];
    const float* w_attn = (const float*)d_in[5];
    const float* b_attn = (const float*)d_in[6];
    const float* ln2_g  = (const float*)d_in[7];
    const float* ln2_b  = (const float*)d_in[8];
    const float* w_fc   = (const float*)d_in[9];
    const float* b_fc   = (const float*)d_in[10];
    const float* w_mlp  = (const float*)d_in[11];
    const float* b_mlp  = (const float*)d_in[12];

    char* ws = (char*)d_ws;
    size_t off = 0;
    auto alloc = [&](size_t bytes) { size_t o = off; off += (bytes + 255) & ~(size_t)255; return o; };
    __bf16* wqkvB = (__bf16*)(ws + alloc((size_t)Cn * 3 * Cn * 2));
    __bf16* wattB = (__bf16*)(ws + alloc((size_t)Cn * Cn * 2));
    __bf16* wfcB  = (__bf16*)(ws + alloc((size_t)Cn * 4 * Cn * 2));
    __bf16* wmlpB = (__bf16*)(ws + alloc((size_t)4 * Cn * Cn * 2));
    __bf16* hb    = (__bf16*)(ws + alloc((size_t)M * Cn * 2));      // LN1 out, reused for LN2 out
    __bf16* bigB  = (__bf16*)(ws + alloc((size_t)M * 4 * Cn * 2));  // qkv (48MB) then fc out (64MB)
    __bf16* yB    = (__bf16*)(ws + alloc((size_t)M * Cn * 2));      // attention out
    float*  x1    = (float*) (ws + alloc((size_t)M * Cn * 4));      // post-attn residual

    __bf16* qkvB = bigB;
    __bf16* fcB  = bigB;

    // weight conversions
    auto cvt = [&](const float* in, __bf16* out, int n) {
        cvt_bf16_kernel<<<(n + 255) / 256, 256, 0, stream>>>(in, out, n);
    };
    cvt(w_qkv,  wqkvB, Cn * 3 * Cn);
    cvt(w_attn, wattB, Cn * Cn);
    cvt(w_fc,   wfcB,  Cn * 4 * Cn);
    cvt(w_mlp,  wmlpB, 4 * Cn * Cn);

    // LN1
    ln_kernel<<<M, 256, 0, stream>>>(x, ln1_g, ln1_b, hb, Cn);

    // QKV GEMM: [M,C] x [C,3C] -> bf16
    {
        dim3 g(3 * Cn / BN, M / BM);
        gemm_bf16_kernel<<<g, 256, 0, stream>>>(hb, wqkvB, b_qkv, nullptr, nullptr, qkvB,
                                                M, 3 * Cn, Cn, 0);
    }
    // Attention
    {
        dim3 g(Bsz * Hn, T / 128);
        attn_kernel<<<g, 256, 0, stream>>>(qkvB, yB, T, Cn, Hn, Dn);
    }
    // Attn proj + residual -> x1 (f32)
    {
        dim3 g(Cn / BN, M / BM);
        gemm_bf16_kernel<<<g, 256, 0, stream>>>(yB, wattB, b_attn, x, x1, nullptr,
                                                M, Cn, Cn, 0);
    }
    // LN2
    ln_kernel<<<M, 256, 0, stream>>>(x1, ln2_g, ln2_b, hb, Cn);

    // FC + GELU -> bf16
    {
        dim3 g(4 * Cn / BN, M / BM);
        gemm_bf16_kernel<<<g, 256, 0, stream>>>(hb, wfcB, b_fc, nullptr, nullptr, fcB,
                                                M, 4 * Cn, Cn, 1);
    }
    // MLP proj + residual -> d_out (f32)
    {
        dim3 g(Cn / BN, M / BM);
        gemm_bf16_kernel<<<g, 256, 0, stream>>>(fcB, wmlpB, b_mlp, x1, (float*)d_out, nullptr,
                                                M, Cn, 4 * Cn, 0);
    }
}
